// _DecoderLSTM_22565758173529
// MI455X (gfx1250) — compile-verified, matrix operand order checked by reading, not run on hardware
//
#include <hip/hip_runtime.h>
#include <hip/hip_bf16.h>

typedef __bf16 bf16_t;
typedef __attribute__((ext_vector_type(16))) __bf16 v16bf;
typedef __attribute__((ext_vector_type(8)))  __bf16 v8bf;
typedef __attribute__((ext_vector_type(8)))  float  v8f;

#define BATCH   64
#define TSTEPS  63      // captions[:, :-1]
#define EDIM    512
#define HDIM    512
#define VDIM    32000
#define G4H     2048    // 4*H
#define MROWS   (TSTEPS * BATCH)   // 4032

#define WMMA_BF16(a, b, c) \
  __builtin_amdgcn_wmma_f32_16x16x32_bf16(false, (a), false, (b), (short)0, (c), false, false)

__device__ __forceinline__ float sigmoidf_(float x) {
  return 1.0f / (1.0f + __expf(-x));
}

// Load one 16x32 bf16 fragment per the CDNA5 VGPR layout:
// lanes 0-15 hold K = {0..7, 16..23}; lanes 16-31 hold K = {8..15, 24..31}.
// lane_row_ptr: this lane's row pointer at the current k offset; khalf = lane>>4.
__device__ __forceinline__ v16bf load_frag16x32(const bf16_t* lane_row_ptr, int khalf) {
  v8bf lo = *(const v8bf*)(lane_row_ptr + (khalf << 3));
  v8bf hi = *(const v8bf*)(lane_row_ptr + 16 + (khalf << 3));
  v16bf r;
#pragma unroll
  for (int i = 0; i < 8; ++i) { r[i] = lo[i]; r[8 + i] = hi[i]; }
  return r;
}

// ---------------------------------------------------------------- conversions
__global__ void k_f32_to_bf16(const float* __restrict__ src, bf16_t* __restrict__ dst, int n) {
  int i = blockIdx.x * blockDim.x + threadIdx.x;
  if (i < n) dst[i] = (bf16_t)src[i];
}

// emb row index r = t*64 + b  (so xp GEMM rows are contiguous per time step)
__global__ void k_embed_gather(const float* __restrict__ table, const int* __restrict__ captions,
                               bf16_t* __restrict__ emb) {
  int r = blockIdx.x;                 // 0 .. 4031
  int t = r >> 6;                     // /64
  int b = r & 63;
  int tok = captions[b * 64 + t];     // captions is (B, T=64); we use t < 63
  int e = threadIdx.x;                // 0 .. 511
  emb[(size_t)r * EDIM + e] = (bf16_t)table[(size_t)tok * EDIM + e];
}

// ------------------------------------------------------------------ h0 / c0
__global__ void k_init_state(const float* __restrict__ enc,
                             const float* __restrict__ Wh, const float* __restrict__ bh,
                             const float* __restrict__ Wc, const float* __restrict__ bc,
                             bf16_t* __restrict__ h0, float* __restrict__ c0) {
  int b = blockIdx.x;                 // 0..63
  int j = threadIdx.x;                // 0..511
  float ah = bh[j], ac = bc[j];
  const float* x  = enc + b * EDIM;
  const float* wh = Wh + (size_t)j * EDIM;
  const float* wc = Wc + (size_t)j * EDIM;
#pragma unroll 4
  for (int e = 0; e < EDIM; ++e) {
    float xv = x[e];
    ah = fmaf(xv, wh[e], ah);
    ac = fmaf(xv, wc[e], ac);
  }
  h0[b * HDIM + j] = (bf16_t)tanhf(ah);
  c0[b * HDIM + j] = tanhf(ac);
}

// -------------------------------------------------- xp = emb @ W_ih^T + bias
// A: (4032 x 512) bf16 row-major, W: (2048 x 512) bf16 row-major, C: (4032 x 2048) f32.
// Block = 256 threads = 8 waves; each wave -> 32x32 C tile (4 WMMAs per 4 frag loads).
__global__ __launch_bounds__(256)
void k_xp_gemm(const bf16_t* __restrict__ A, const bf16_t* __restrict__ W,
               const float* __restrict__ b_ih, const float* __restrict__ b_hh,
               float* __restrict__ C) {
  int wave = threadIdx.x >> 5;
  int lane = threadIdx.x & 31;
  int khalf = lane >> 4;
  int l15 = lane & 15;
  int row_base = blockIdx.y * 32;
  int n0 = blockIdx.x * 256 + wave * 32;

  const bf16_t* a_row0 = A + (size_t)(row_base + l15) * EDIM;
  const bf16_t* a_row1 = A + (size_t)(row_base + 16 + l15) * EDIM;
  const bf16_t* b_row0 = W + (size_t)(n0 + l15) * EDIM;
  const bf16_t* b_row1 = W + (size_t)(n0 + 16 + l15) * EDIM;

  v8f acc00 = {}, acc01 = {}, acc10 = {}, acc11 = {};
  for (int k = 0; k < EDIM; k += 32) {
    v16bf a0 = load_frag16x32(a_row0 + k, khalf);
    v16bf a1 = load_frag16x32(a_row1 + k, khalf);
    v16bf w0 = load_frag16x32(b_row0 + k, khalf);
    v16bf w1 = load_frag16x32(b_row1 + k, khalf);
    acc00 = WMMA_BF16(a0, w0, acc00);
    acc01 = WMMA_BF16(a0, w1, acc01);
    acc10 = WMMA_BF16(a1, w0, acc10);
    acc11 = WMMA_BF16(a1, w1, acc11);
  }
  int c0i = n0 + l15;
  int c1i = n0 + 16 + l15;
  float bias0 = b_ih[c0i] + b_hh[c0i];
  float bias1 = b_ih[c1i] + b_hh[c1i];
#pragma unroll
  for (int j = 0; j < 8; ++j) {
    int m = j + (khalf << 3);
    size_t r0 = (size_t)(row_base + m) * G4H;
    size_t r1 = (size_t)(row_base + 16 + m) * G4H;
    C[r0 + c0i] = acc00[j] + bias0;
    C[r0 + c1i] = acc01[j] + bias1;
    C[r1 + c0i] = acc10[j] + bias0;
    C[r1 + c1i] = acc11[j] + bias1;
  }
}

// ----------------------------------------------------------- one LSTM step
// gates = xp_t + h @ W_hh^T ; each wave owns 16 batch rows x 16 h-cols and all 4 gates
// (one A fragment feeds 4 gate WMMAs). Grid: (HDIM/64, BATCH/16), block = 128 = 4 waves.
__global__ __launch_bounds__(128)
void k_lstm_step(const float* __restrict__ xp_t,      // 64 x 2048 (this step)
                 const bf16_t* __restrict__ Whh,      // 2048 x 512 bf16
                 const bf16_t* __restrict__ h_in,     // 64 x 512 bf16
                 bf16_t* __restrict__ h_out,          // 64 x 512 bf16 (ping-pong)
                 bf16_t* __restrict__ hs,             // 4032 x 512 bf16, row = b*63+t
                 int t,
                 float* __restrict__ c) {             // 64 x 512 f32 (in/out)
  int wave = threadIdx.x >> 5;
  int lane = threadIdx.x & 31;
  int khalf = lane >> 4;
  int l15 = lane & 15;
  int row_base = blockIdx.y * 16;
  int n0 = blockIdx.x * 64 + wave * 16;

  v8f acc[4];
#pragma unroll
  for (int q = 0; q < 4; ++q) {
#pragma unroll
    for (int j = 0; j < 8; ++j) {
      int row = row_base + j + (khalf << 3);
      acc[q][j] = xp_t[(size_t)row * G4H + q * HDIM + n0 + l15];
    }
  }

  const bf16_t* a_row = h_in + (size_t)(row_base + l15) * HDIM;
  for (int k = 0; k < HDIM; k += 32) {
    v16bf a = load_frag16x32(a_row + k, khalf);      // shared by all 4 gate WMMAs
#pragma unroll
    for (int q = 0; q < 4; ++q) {
      const bf16_t* b_row = Whh + (size_t)(q * HDIM + n0 + l15) * HDIM + k;
      v16bf bq = load_frag16x32(b_row, khalf);
      acc[q] = WMMA_BF16(a, bq, acc[q]);
    }
  }

#pragma unroll
  for (int j = 0; j < 8; ++j) {
    int row = row_base + j + (khalf << 3);           // batch index b
    int col = n0 + l15;
    float iv = sigmoidf_(acc[0][j]);
    float fv = sigmoidf_(acc[1][j]);
    float gv = tanhf(acc[2][j]);
    float ov = sigmoidf_(acc[3][j]);
    size_t idx = (size_t)row * HDIM + col;
    float cn = fv * c[idx] + iv * gv;
    float hn = ov * tanhf(cn);
    c[idx] = cn;
    bf16_t hb = (bf16_t)hn;
    h_out[idx] = hb;                                  // ping-pong (t-major use)
    hs[((size_t)row * TSTEPS + t) * HDIM + col] = hb; // output-row order b*63+t
  }
}

// ------------------------------------------------- logits = hs @ Wout^T + b
// hs already in output-row order, so this is a plain row-major GEMM.
// Block = 256 threads = 8 waves; each wave -> 32x32 tile: 4 frag loads -> 4 WMMAs.
__global__ __launch_bounds__(256)
void k_logits_gemm(const bf16_t* __restrict__ hs,     // 4032 x 512, row = b*63+t
                   const bf16_t* __restrict__ Wout,   // 32000 x 512 bf16
                   const float* __restrict__ bout,
                   float* __restrict__ out) {         // 4032 x 32000 f32
  int wave = threadIdx.x >> 5;
  int lane = threadIdx.x & 31;
  int khalf = lane >> 4;
  int l15 = lane & 15;
  int row_base = blockIdx.y * 32;
  int n0 = blockIdx.x * 256 + wave * 32;

  const bf16_t* a_row0 = hs + (size_t)(row_base + l15) * HDIM;
  const bf16_t* a_row1 = hs + (size_t)(row_base + 16 + l15) * HDIM;
  const bf16_t* b_row0 = Wout + (size_t)(n0 + l15) * HDIM;
  const bf16_t* b_row1 = Wout + (size_t)(n0 + 16 + l15) * HDIM;

  v8f acc00 = {}, acc01 = {}, acc10 = {}, acc11 = {};
  for (int k = 0; k < HDIM; k += 32) {
    if (k + 64 < HDIM) __builtin_prefetch(b_row0 + k + 64, 0, 1);
    v16bf a0 = load_frag16x32(a_row0 + k, khalf);
    v16bf a1 = load_frag16x32(a_row1 + k, khalf);
    v16bf w0 = load_frag16x32(b_row0 + k, khalf);
    v16bf w1 = load_frag16x32(b_row1 + k, khalf);
    acc00 = WMMA_BF16(a0, w0, acc00);
    acc01 = WMMA_BF16(a0, w1, acc01);
    acc10 = WMMA_BF16(a1, w0, acc10);
    acc11 = WMMA_BF16(a1, w1, acc11);
  }
  int c0i = n0 + l15;
  int c1i = n0 + 16 + l15;
  float bias0 = bout[c0i];
  float bias1 = bout[c1i];
#pragma unroll
  for (int j = 0; j < 8; ++j) {
    int m = j + (khalf << 3);
    size_t r0 = (size_t)(row_base + m) * VDIM;
    size_t r1 = (size_t)(row_base + 16 + m) * VDIM;
    out[r0 + c0i] = acc00[j] + bias0;
    out[r0 + c1i] = acc01[j] + bias1;
    out[r1 + c0i] = acc10[j] + bias0;
    out[r1 + c1i] = acc11[j] + bias1;
  }
}

// ---------------------------------------------------------------- launcher
extern "C" void kernel_launch(void* const* d_in, const int* in_sizes, int n_in,
                              void* d_out, int out_size, void* d_ws, size_t ws_size,
                              hipStream_t stream) {
  const float* enc      = (const float*)d_in[0];
  const int*   captions = (const int*)  d_in[1];
  const float* table    = (const float*)d_in[2];
  const float* W_ih     = (const float*)d_in[3];
  const float* W_hh     = (const float*)d_in[4];
  const float* b_ih     = (const float*)d_in[5];
  const float* b_hh     = (const float*)d_in[6];
  const float* Wh       = (const float*)d_in[7];
  const float* bh       = (const float*)d_in[8];
  const float* Wc       = (const float*)d_in[9];
  const float* bc       = (const float*)d_in[10];
  const float* Wout     = (const float*)d_in[11];
  const float* bout     = (const float*)d_in[12];
  float* out = (float*)d_out;

  // workspace carve-up (256B aligned)
  char* base = (char*)d_ws;
  size_t off = 0;
  auto carve = [&](size_t bytes) -> void* {
    void* p = base + off;
    off = (off + bytes + 255) & ~(size_t)255;
    return p;
  };
  bf16_t* Wih_b  = (bf16_t*)carve((size_t)G4H * EDIM * sizeof(bf16_t));    //  2 MB
  bf16_t* Whh_b  = (bf16_t*)carve((size_t)G4H * HDIM * sizeof(bf16_t));    //  2 MB
  bf16_t* Wout_b = (bf16_t*)carve((size_t)VDIM * HDIM * sizeof(bf16_t));   // 33 MB
  bf16_t* emb_b  = (bf16_t*)carve((size_t)MROWS * EDIM * sizeof(bf16_t));  //  4 MB
  float*  xp     = (float*) carve((size_t)MROWS * G4H * sizeof(float));    // 33 MB
  float*  cstate = (float*) carve((size_t)BATCH * HDIM * sizeof(float));
  bf16_t* hbuf0  = (bf16_t*)carve((size_t)BATCH * HDIM * sizeof(bf16_t));
  bf16_t* hbuf1  = (bf16_t*)carve((size_t)BATCH * HDIM * sizeof(bf16_t));
  bf16_t* hs     = (bf16_t*)carve((size_t)MROWS * HDIM * sizeof(bf16_t));  //  4 MB
  (void)ws_size; (void)in_sizes; (void)n_in; (void)out_size;

  // 1) weight conversions to bf16
  {
    int n = G4H * EDIM;
    k_f32_to_bf16<<<(n + 255) / 256, 256, 0, stream>>>(W_ih, Wih_b, n);
    k_f32_to_bf16<<<(n + 255) / 256, 256, 0, stream>>>(W_hh, Whh_b, n);
    int nv = VDIM * HDIM;
    k_f32_to_bf16<<<(nv + 255) / 256, 256, 0, stream>>>(Wout, Wout_b, nv);
  }

  // 2) embedding gather (bf16, row = t*64+b)
  k_embed_gather<<<MROWS, EDIM, 0, stream>>>(table, captions, emb_b);

  // 3) h0 (bf16 into ping-pong buf 0) and c0 (f32)
  k_init_state<<<BATCH, HDIM, 0, stream>>>(enc, Wh, bh, Wc, bc, hbuf0, cstate);

  // 4) xp = emb @ W_ih^T + (b_ih + b_hh)
  k_xp_gemm<<<dim3(G4H / 256, MROWS / 32), 256, 0, stream>>>(emb_b, Wih_b, b_ih, b_hh, xp);

  // 5) recurrence: 63 sequential steps, ping-pong h buffers, hs in output-row order
  bf16_t* hb[2] = {hbuf0, hbuf1};
  for (int t = 0; t < TSTEPS; ++t) {
    k_lstm_step<<<dim3(HDIM / 64, BATCH / 16), 128, 0, stream>>>(
        xp + (size_t)t * BATCH * G4H, Whh_b,
        hb[t & 1], hb[(t + 1) & 1],
        hs, t, cstate);
  }

  // 6) logits = hs @ Wout^T + bout
  k_logits_gemm<<<dim3(VDIM / 256, MROWS / 32), 256, 0, stream>>>(hs, Wout_b, bout, out);
}
